// LinearChainCRF_5849745457169
// MI455X (gfx1250) — matrix operand order repaired and checked by reading, not compiled, not run on hardware
//
#include <hip/hip_runtime.h>
#include <cstdint>
#include <cstddef>

#define TT 1024
#define CC 256
#define BB 64
#define CBIAS 8.0f   // f16 overflow guard bias folded into p, re-added in log space

typedef __attribute__((ext_vector_type(16))) _Float16 v16h;
typedef __attribute__((ext_vector_type(8)))  _Float16 v8h;
typedef __attribute__((ext_vector_type(8)))  float    v8f;
typedef __attribute__((ext_vector_type(4)))  unsigned int u32x4;
typedef __attribute__((ext_vector_type(8)))  int      i32x8;
typedef __attribute__((ext_vector_type(4)))  int      i32x4;

// Raw transcendentals: inputs are always finite normals here, so skip the
// libm denorm/range fixups. v_nop satisfies the ISA TRANS-op hazard rule.
__device__ __forceinline__ float fast_exp(float x) {
  float r;
  const float xs = x * 1.44269504088896341f;   // 1/ln2
  asm("v_exp_f32 %0, %1\n\tv_nop" : "=v"(r) : "v"(xs));
  return r;
}
__device__ __forceinline__ float fast_log(float x) {
  float r;
  asm("v_log_f32 %0, %1\n\tv_nop" : "=v"(r) : "v"(x));
  return r * 0.69314718055994531f;             // ln2
}

__device__ __forceinline__ unsigned lds_off(const void* p) {
  // generic pointer to LDS: low 32 bits are the LDS byte address
  return (unsigned)(unsigned long long)(uintptr_t)p;
}

// TDM: load one row of CC f32 (1KB) from global into LDS, tracked by TENSORcnt.
__device__ __forceinline__ void tdm_load_row(unsigned lds_byte, const float* g) {
  unsigned long long ga = (unsigned long long)(uintptr_t)g;
  u32x4 g0;
  g0[0] = 1u;                                   // count=1, is_restore=0, gather=0
  g0[1] = lds_byte;                             // lds_addr
  g0[2] = (unsigned)ga;                         // global_addr[31:0]
  g0[3] = (unsigned)((ga >> 32) & 0x01ffffffull) | (2u << 30); // addr[56:32], type=2
  i32x8 g1;
  g1[0] = 0x00020000;                           // wg_mask=0, data_size=2 (4B)
  g1[1] = (int)((unsigned)CC << 16);            // tensor_dim0 low16
  g1[2] = (int)(1u << 16);                      // tensor_dim0 hi=0, tensor_dim1=1 (low16)
  g1[3] = (int)((unsigned)CC << 16);            // tensor_dim1 hi=0, tile_dim0=CC
  g1[4] = 1;                                    // tile_dim1=1, tile_dim2=0
  g1[5] = CC;                                   // tensor_dim0_stride low32
  g1[6] = 0;
  g1[7] = 0;
  i32x4 z4 = {0, 0, 0, 0};
#if defined(__clang_major__) && (__clang_major__ >= 23)
  i32x8 z8 = {0, 0, 0, 0, 0, 0, 0, 0};
  __builtin_amdgcn_tensor_load_to_lds(g0, g1, z4, z4, z8, 0);
#else
  __builtin_amdgcn_tensor_load_to_lds(g0, g1, z4, z4, 0);
#endif
}

__device__ __forceinline__ float redmax16(float v) {
#pragma unroll
  for (int d = 1; d < 16; d <<= 1) v = fmaxf(v, __shfl_xor(v, d));
  return v;
}

__global__ void __launch_bounds__(512)
crf_forward_kernel(const float* __restrict__ emis,
                   const float* __restrict__ trans,
                   const int*   __restrict__ seq_lens,
                   float*       __restrict__ out) {
  __shared__ __align__(16) _Float16 pbuf[2][CC];   // p = exp(alpha - m - c), WMMA-A src
  __shared__ __align__(16) float    ebuf[2][CC];   // TDM emission double buffer
  __shared__ float wmaxbuf[2][16];                 // per-wave alpha maxes (lag-1)
  __shared__ float finalbuf[CC];

  const int tid  = threadIdx.x;
  const int wave = tid >> 5;          // 0..15 -> owns j-chunk [16*wave, 16*wave+16)
  const int lane = tid & 31;
  const int half = lane >> 4;         // 0 or 1 (WMMA half-wave)
  const int jcol = (wave << 4) + (lane & 15);
  const int b    = blockIdx.x;
  const int len  = seq_lens[b];

  const float* eb   = emis  + (size_t)b * TT * CC;
  const float* tb   = trans + (size_t)b * CC * CC;
  float*       outA = out   + (size_t)b * TT * CC;
  float*       outZ = out   + (size_t)BB * TT * CC;

  // ---- E = exp(trans_b), register-resident in WMMA-B (32x16 f16) layout ----
  // breg[kc] element r: K = 32*kc + (half ? 16+r : r), N = jcol
  v16h breg[8];
#pragma unroll
  for (int kc = 0; kc < 8; ++kc) {
    const int ibase = 32 * kc + (half ? 16 : 0);
#pragma unroll
    for (int r = 0; r < 16; ++r) {
      breg[kc][r] = (_Float16)fast_exp(tb[(ibase + r) * CC + jcol]);
    }
  }

  // ---- prologue: TDM-prefetch emission rows 0 and 1 ----
  if (wave == 0) {
    tdm_load_row(lds_off(&ebuf[0][0]), eb);
    tdm_load_row(lds_off(&ebuf[1][0]), eb + CC);
    __builtin_amdgcn_s_wait_tensorcnt(0);
  }
  __syncthreads();

  // ---- t = 0: alpha = emis[0] ----
  float alpha = ebuf[0][jcol];
  if (lane < 16) {
    outA[jcol] = alpha;
    if (len == 1) finalbuf[jcol] = alpha;
  }
  {
    float wm = redmax16(alpha);
    if (lane == 0) wmaxbuf[0][wave] = wm;        // slot read by step t=1
  }
  __syncthreads();                               // prologue-only extra barrier
  float mv = wmaxbuf[0][lane & 15];
  float mreg = redmax16(mv);                     // exact max(alpha_0)
  {
    const float p = fast_exp(alpha - mreg - CBIAS);
    if (lane < 16) pbuf[1][jcol] = (_Float16)p;  // consumed by step t=1
  }
  __syncthreads();

  // ---- sequential scan: ONE barrier per step (lag-1 max rescaling) ----
  // Entry invariants at step t: pbuf[t&1] = exp(alpha_{t-1} - mreg - c);
  // ebuf[t&1] = emis[t]; wmaxbuf[(t-1)&1] = wave-maxes of alpha_{t-1}.
#pragma unroll 1
  for (int t = 1; t < TT; ++t) {
    const int cur = t & 1;
    const int nxt = cur ^ 1;
    if (wave == 0 && (t + 1) < TT) {
      tdm_load_row(lds_off(&ebuf[nxt][0]), eb + (size_t)(t + 1) * CC);
    }

    // Issue the lag-1 max source load FIRST (DS retires in order), then the
    // 16 A-chunk loads; the bpermute tree for m_new runs under the WMMA chain.
    const float mwv = wmaxbuf[nxt][lane & 15];

    const v8h* pb = (const v8h*)&pbuf[cur][0];
    v16h areg[8];
#pragma unroll
    for (int kc = 0; kc < 8; ++kc) {
      v8h lo = pb[4 * kc + 2 * half];      // K = 32kc + 16*half + {0..7}
      v8h hi = pb[4 * kc + 2 * half + 1];  // K = 32kc + 16*half + {8..15}
      areg[kc] = __builtin_shufflevector(lo, hi, 0, 1, 2, 3, 4, 5, 6, 7,
                                                 8, 9, 10, 11, 12, 13, 14, 15);
    }
    const float ecur = ebuf[cur][jcol];

    const float m_new = redmax16(mwv);     // max(alpha_{t-1}), off the tail

    v8f acc = {0.f, 0.f, 0.f, 0.f, 0.f, 0.f, 0.f, 0.f};
#pragma unroll
    for (int kc = 0; kc < 8; ++kc) {
      acc = __builtin_amdgcn_wmma_f32_16x16x32_f16(false, areg[kc], false, breg[kc],
                                                   (short)0, acc, false, false);
    }
    const float s = acc[0];                // all D rows identical (A rows replicated)

    alpha = ecur + mreg + CBIAS + fast_log(s);
    if (lane < 16) {
      outA[(size_t)t * CC + jcol] = alpha;
      if (t == len - 1) finalbuf[jcol] = alpha;
    }

    const float p = fast_exp(alpha - m_new - CBIAS);
    if (lane < 16) pbuf[nxt][jcol] = (_Float16)p;

    // publish this step's wave-max for step t+1
    float wm = redmax16(alpha);
    if (lane == 0) wmaxbuf[cur][wave] = wm;

    mreg = m_new;
    if (wave == 0) __builtin_amdgcn_s_wait_tensorcnt(0);  // emis[t+1] landed
    __syncthreads();
  }

  // ---- log_Z = logsumexp_j(alpha[len-1, j]) ----
  if (wave == 0) {
    float v[8];
    float lm = -3.402823466e38f;
#pragma unroll
    for (int k = 0; k < 8; ++k) {
      v[k] = finalbuf[lane * 8 + k];
      lm = fmaxf(lm, v[k]);
    }
#pragma unroll
    for (int d = 1; d < 32; d <<= 1) lm = fmaxf(lm, __shfl_xor(lm, d));
    float ssum = 0.f;
#pragma unroll
    for (int k = 0; k < 8; ++k) ssum += fast_exp(v[k] - lm);
#pragma unroll
    for (int d = 1; d < 32; d <<= 1) ssum += __shfl_xor(ssum, d);
    if (lane == 0) outZ[b] = lm + fast_log(ssum);
  }
}

extern "C" void kernel_launch(void* const* d_in, const int* in_sizes, int n_in,
                              void* d_out, int out_size, void* d_ws, size_t ws_size,
                              hipStream_t stream) {
  const float* emis     = (const float*)d_in[0];   // (B,T,C) f32
  const float* trans    = (const float*)d_in[1];   // (B,C,C) f32
  const int*   seq_lens = (const int*)d_in[2];     // (B,)    i32
  float*       out      = (float*)d_out;           // alpha (B*T*C) ++ log_Z (B)
  (void)in_sizes; (void)n_in; (void)out_size; (void)d_ws; (void)ws_size;
  crf_forward_kernel<<<BB, 512, 0, stream>>>(emis, trans, seq_lens, out);
}
